// EMO_Router_Scaffold_32899449487922
// MI455X (gfx1250) — matrix operand order repaired and checked by reading, not compiled, not run on hardware
//
#include <hip/hip_runtime.h>

// ---------------------------------------------------------------------------
// Fused EMO-router: logits = x @ W_gate^T, softmax, segment stats, penalty.
//   x:           (262144, 1024) f32
//   doc_indices: (262144,)      int
//   W_gate:      (64, 1024)     f32
// Outputs: w (262144,64) f32  then  doc_penalty scalar.
//
// GEMM runs on the WMMA pipe via v_wmma_f32_16x16x32_f16 with a 2-term f16
// split of both operands (3 WMMAs: hi*hi + hi*lo + lo*hi) for fp32-class
// accuracy.  W_gate is staged per-128-K-chunk into LDS as f16 hi/lo with a
// padded stride for conflict-free ds_load_b128.  Softmax + per-doc atomics
// are fused in-register; tables live in L2.
// ---------------------------------------------------------------------------

#define N_ROWS   262144
#define D_DIM    1024
#define E_DIM    64
#define NDOCS    1024
#define KC       128        // K-chunk staged in LDS
#define KCP      136        // padded LDS row stride (halves): 272B = 68 dwords
                            //  -> 4-bank rotation per e-row, conflict-free

typedef _Float16 v16h __attribute__((ext_vector_type(16)));
typedef _Float16 v8h  __attribute__((ext_vector_type(8)));
typedef float    v8f  __attribute__((ext_vector_type(8)));
typedef float    v4f  __attribute__((ext_vector_type(4)));

union V16H { v16h v; v8h h2[2]; _Float16 e[16]; };

__global__ __launch_bounds__(256)
void emo_router_gemm_softmax(const float* __restrict__ x,
                             const int*   __restrict__ docs,
                             const float* __restrict__ Wg,
                             float*       __restrict__ w_out,
                             float*       __restrict__ sumw,
                             float*       __restrict__ sumw2,
                             float*       __restrict__ counts)
{
    __shared__ _Float16 Whi[E_DIM * KCP];   // 17408 B
    __shared__ _Float16 Wlo[E_DIM * KCP];   // 17408 B  (total ~34 KB LDS)

    const int tid  = threadIdx.x;
    const int l    = tid & 31;          // lane in wave32
    const int wave = tid >> 5;          // 8 waves / block
    const int half = l >> 4;            // half-wave id (ISA A/B layout split)
    const int ln   = l & 15;
    const int rowBase = (blockIdx.x * 8 + wave) * 16;   // wave's 16-row tile
    const int myRow   = rowBase + ln;                    // row this lane loads

    v8f acc[4];                          // 16x64 f32 logit tile per wave
    #pragma unroll
    for (int t = 0; t < 4; ++t)
        #pragma unroll
        for (int i = 0; i < 8; ++i) acc[t][i] = 0.0f;

    for (int kc = 0; kc < D_DIM; kc += KC) {
        __syncthreads();                 // protect LDS from previous chunk
        // Stage W chunk -> LDS as f16 hi/lo (converted once per block/chunk)
        for (int i = tid; i < E_DIM * KC; i += 256) {
            const int e = i >> 7;        // / KC
            const int d = i & (KC - 1);
            const float v = Wg[e * D_DIM + kc + d];
            const _Float16 h = (_Float16)v;
            Whi[e * KCP + d] = h;
            Wlo[e * KCP + d] = (_Float16)(v - (float)h);
        }
        __syncthreads();

        if (kc + KC < D_DIM)             // warm L2/L0 for next x chunk
            __builtin_prefetch(&x[(size_t)myRow * D_DIM + kc + KC + half * 64], 0, 1);

        #pragma unroll
        for (int ks = 0; ks < KC / 32; ++ks) {
            const int k0 = kc + ks * 32;
            // ---- A fragment (16x32 f16 layout): lanes 0-15 hold K 0-7 &
            //      16-23, lanes 16-31 hold K 8-15 & 24-31, row = lane&15.
            const float* xr = x + (size_t)myRow * D_DIM + k0 + half * 8;
            const v4f q0 = *(const v4f*)(xr);
            const v4f q1 = *(const v4f*)(xr + 4);
            const v4f q2 = *(const v4f*)(xr + 16);
            const v4f q3 = *(const v4f*)(xr + 20);
            float af[16];
            #pragma unroll
            for (int i = 0; i < 4; ++i) {
                af[i] = q0[i]; af[4 + i] = q1[i];
                af[8 + i] = q2[i]; af[12 + i] = q3[i];
            }
            V16H ahi, alo;
            #pragma unroll
            for (int i = 0; i < 16; ++i) {
                const _Float16 h = (_Float16)af[i];
                ahi.e[i] = h;
                alo.e[i] = (_Float16)(af[i] - (float)h);
            }

            // ---- B fragments (32x16): lane = column e, same K split as A
            //      (K 8h..8h+7 & 16+8h..23+8h).  3-term f16 emulation of f32 GEMM.
            #pragma unroll
            for (int t = 0; t < 4; ++t) {
                const _Float16* bh = &Whi[(t * 16 + ln) * KCP + ks * 32 + half * 8];
                const _Float16* bl = &Wlo[(t * 16 + ln) * KCP + ks * 32 + half * 8];
                V16H Bh, Bl;
                Bh.h2[0] = *(const v8h*)(bh);
                Bh.h2[1] = *(const v8h*)(bh + 16);
                Bl.h2[0] = *(const v8h*)(bl);
                Bl.h2[1] = *(const v8h*)(bl + 16);
                acc[t] = __builtin_amdgcn_wmma_f32_16x16x32_f16(
                             false, ahi.v, false, Bh.v, (short)0, acc[t], false, false);
                acc[t] = __builtin_amdgcn_wmma_f32_16x16x32_f16(
                             false, ahi.v, false, Bl.v, (short)0, acc[t], false, false);
                acc[t] = __builtin_amdgcn_wmma_f32_16x16x32_f16(
                             false, alo.v, false, Bh.v, (short)0, acc[t], false, false);
            }
        }
    }

    // ------------------------------------------------------------------
    // Softmax over E=64 per row + fused segment-sum atomics.
    // C layout: VGPR r, lanes 0-15 -> row r, lanes 16-31 -> row 8+r,
    // column = lane&15 (+16*t).  Row reductions = half-wave shfl_xor.
    // ------------------------------------------------------------------
    #pragma unroll
    for (int r = 0; r < 8; ++r) {
        float v0 = acc[0][r], v1 = acc[1][r], v2 = acc[2][r], v3 = acc[3][r];
        float mx = fmaxf(fmaxf(v0, v1), fmaxf(v2, v3));
        #pragma unroll
        for (int m = 1; m < 16; m <<= 1) mx = fmaxf(mx, __shfl_xor(mx, m, 32));
        float e0 = __expf(v0 - mx), e1 = __expf(v1 - mx);
        float e2 = __expf(v2 - mx), e3 = __expf(v3 - mx);
        float s = e0 + e1 + e2 + e3;
        #pragma unroll
        for (int m = 1; m < 16; m <<= 1) s += __shfl_xor(s, m, 32);
        const float inv = 1.0f / s;
        e0 *= inv; e1 *= inv; e2 *= inv; e3 *= inv;

        const int row = rowBase + r + half * 8;
        float* op = w_out + (size_t)row * E_DIM + ln;
        op[0] = e0; op[16] = e1; op[32] = e2; op[48] = e3;

        const int doc = docs[row];
        float* swp  = sumw  + doc * E_DIM + ln;
        float* sw2p = sumw2 + doc * E_DIM + ln;
        unsafeAtomicAdd(swp + 0,  e0);       unsafeAtomicAdd(sw2p + 0,  e0 * e0);
        unsafeAtomicAdd(swp + 16, e1);       unsafeAtomicAdd(sw2p + 16, e1 * e1);
        unsafeAtomicAdd(swp + 32, e2);       unsafeAtomicAdd(sw2p + 32, e2 * e2);
        unsafeAtomicAdd(swp + 48, e3);       unsafeAtomicAdd(sw2p + 48, e3 * e3);
        if (ln == 0) unsafeAtomicAdd(&counts[doc], 1.0f);
    }
}

// ---------------------------------------------------------------------------
// Fold 1024-doc tables into the scalar penalty.
// ---------------------------------------------------------------------------
__global__ __launch_bounds__(1024)
void emo_router_finalize(const float* __restrict__ sumw,
                         const float* __restrict__ sumw2,
                         const float* __restrict__ counts,
                         float*       __restrict__ out_penalty)
{
    __shared__ float sp[1024];
    __shared__ float sc[1024];
    const int d = threadIdx.x;

    const float n   = counts[d];
    const float sn  = fmaxf(n, 1.0f);
    const float inv = 1.0f / sn;
    float ssd = 0.0f;
    #pragma unroll 8
    for (int e = 0; e < E_DIM; ++e) {
        const float s  = sumw [d * E_DIM + e];
        const float s2 = sumw2[d * E_DIM + e];
        ssd += s2 - s * s * inv;           // sum_w2 - safe_n*centroid^2
    }
    const float mse = ssd * inv * (1.0f / (float)E_DIM);
    sp[d] = (n > 1.0f) ? mse : 0.0f;
    sc[d] = (n > 0.0f) ? 1.0f : 0.0f;
    __syncthreads();
    for (int st = 512; st > 0; st >>= 1) {
        if (d < st) { sp[d] += sp[d + st]; sc[d] += sc[d + st]; }
        __syncthreads();
    }
    if (d == 0) out_penalty[0] = sp[0] / sc[0];
}

extern "C" void kernel_launch(void* const* d_in, const int* in_sizes, int n_in,
                              void* d_out, int out_size, void* d_ws, size_t ws_size,
                              hipStream_t stream)
{
    const float* x    = (const float*)d_in[0];
    const int*   docs = (const int*)  d_in[1];   // harness: integer -> int*
    const float* Wg   = (const float*)d_in[2];
    float* out = (float*)d_out;

    float* sumw   = (float*)d_ws;                        // 1024*64
    float* sumw2  = sumw  + NDOCS * E_DIM;               // 1024*64
    float* counts = sumw2 + NDOCS * E_DIM;               // 1024
    hipMemsetAsync(d_ws, 0, (size_t)(2 * NDOCS * E_DIM + NDOCS) * sizeof(float),
                   stream);

    const int blocks = N_ROWS / 128;     // 8 waves * 16 rows per block
    emo_router_gemm_softmax<<<blocks, 256, 0, stream>>>(
        x, docs, Wg, out, sumw, sumw2, counts);
    emo_router_finalize<<<1, 1024, 0, stream>>>(
        sumw, sumw2, counts, out + (size_t)N_ROWS * E_DIM);
}
